// GNNRegressor_39402029973935
// MI455X (gfx1250) — matrix operand order repaired and checked by reading, not compiled, not run on hardware
//
#include <hip/hip_runtime.h>
#include <hip/hip_bf16.h>

typedef __attribute__((ext_vector_type(16))) _Float16 v16h;
typedef __attribute__((ext_vector_type(8)))  float    v8f;

#define NGRAPH_MAX 256

// ---------------------------------------------------------------------------
// ws layout (floats):
//   dinv   : N
//   xs     : N      (phase 1: deg (seeded 1.0 for self loop); phase 2: x*dinv)
//   asum   : N      (layer-1 scalar aggregation)
//   gsum   : 256
//   gcnt   : 256
//   h1s    : N*32   (h1 * dinv  -> source-side scaled messages)
//   aggsum : N*32   (seeded with h1s[i]; final agg = aggsum * dinv)
// ---------------------------------------------------------------------------

__global__ void k_init(float* __restrict__ deg, float* __restrict__ asum,
                       float* __restrict__ gsum, float* __restrict__ gcnt, int N) {
  int i = blockIdx.x * blockDim.x + threadIdx.x;
  if (i < N) { deg[i] = 1.0f; asum[i] = 0.0f; }      // 1.0 = self-loop
  if (i < NGRAPH_MAX) { gsum[i] = 0.0f; gcnt[i] = 0.0f; }
}

__global__ void k_deg(const int* __restrict__ dst, float* __restrict__ deg, int E) {
  int e = blockIdx.x * blockDim.x + threadIdx.x;
  if (e < E) atomicAdd(&deg[dst[e]], 1.0f);
}

// deg -> dinv, xs = x*dinv (reuses deg buffer), and per-graph node counts
__global__ void k_dinv(const float* __restrict__ x, const int* __restrict__ batch,
                       float* __restrict__ degxs, float* __restrict__ dinv,
                       float* __restrict__ gcnt, int N) {
  int i = blockIdx.x * blockDim.x + threadIdx.x;
  if (i >= N) return;
  float dv = rsqrtf(degxs[i]);     // deg >= 1 always (self loop)
  dinv[i]  = dv;
  degxs[i] = x[i] * dv;            // xs
  atomicAdd(&gcnt[batch[i]], 1.0f);
}

// layer-1 scalar edge scatter: asum[d] += x[s]*dinv[s]
__global__ void k_edge1(const int* __restrict__ src, const int* __restrict__ dst,
                        const float* __restrict__ xs, float* __restrict__ asum, int E) {
  int e = blockIdx.x * blockDim.x + threadIdx.x;
  if (e < E) atomicAdd(&asum[dst[e]], xs[src[e]]);
}

// per (node, k): h1 = relu(a*W1+b1); h1s = h1*dinv; seed aggsum with h1s
__global__ void k_node1(const float* __restrict__ asum, const float* __restrict__ xs,
                        const float* __restrict__ dinv, const float* __restrict__ W1,
                        const float* __restrict__ b1, float* __restrict__ h1s,
                        float* __restrict__ aggsum, int N) {
  int t = blockIdx.x * blockDim.x + threadIdx.x;
  int i = t >> 5, k = t & 31;
  if (i >= N) return;
  float dv = dinv[i];
  float a  = (asum[i] + xs[i]) * dv;             // edges + self-loop, * dinv[dst]
  float h  = fmaxf(a * W1[k] + b1[k], 0.0f);
  float hs = h * dv;
  h1s[i * 32 + k]    = hs;
  aggsum[i * 32 + k] = hs;                       // self-loop seed (final *dv -> h*dv^2)
}

// layer-2 edge scatter, 32 channels/edge: aggsum[d][k] += h1s[s][k]
__global__ void k_edge2(const int* __restrict__ src, const int* __restrict__ dst,
                        const float* __restrict__ h1s, float* __restrict__ aggsum,
                        int E32) {
  int t = blockIdx.x * blockDim.x + threadIdx.x;
  if (t >= E32) return;
  int e = t >> 5, k = t & 31;
  atomicAdd(&aggsum[dst[e] * 32 + k], h1s[src[e] * 32 + k]);
}

// One wave per 16-node tile: agg[16,32](f16) x W2[32,64](f16) via 4x
// v_wmma_f32_16x16x32_f16, fused bias+relu+(.Wl), lane-reduce, per-graph atomic.
__global__ void k_gemm_pool(const float* __restrict__ aggsum, const float* __restrict__ dinv,
                            const float* __restrict__ W2, const float* __restrict__ b2,
                            const float* __restrict__ Wl, const int* __restrict__ batch,
                            float* __restrict__ gsum, int N) {
  int wave = (blockIdx.x * blockDim.x + threadIdx.x) >> 5;
  int lane = threadIdx.x & 31;
  int ntiles = (N + 15) >> 4;
  if (wave >= ntiles) return;                    // uniform per wave: EXEC stays full

  int m  = lane & 15;                            // row within tile / col within B tile
  int kh = lane >> 4;                            // lane-half
  int node = wave * 16 + m;
  int nclamp = node < N ? node : (N - 1);        // keep EXEC all-ones on ragged tail

  float dv = dinv[nclamp];
  const float* arow = aggsum + nclamp * 32;

  // A: 16x32 f16, documented CDNA5 layout
  v16h A;
#pragma unroll
  for (int j = 0; j < 8; ++j) {
    A[j]     = (_Float16)(arow[kh * 8 + j]      * dv);
    A[j + 8] = (_Float16)(arow[16 + kh * 8 + j] * dv);
  }

  float p[8];
#pragma unroll
  for (int r = 0; r < 8; ++r) p[r] = 0.0f;

#pragma unroll
  for (int t = 0; t < 4; ++t) {                  // output cols 16t .. 16t+15
    v16h B;                                      // B: 32x16 f16, K = 8*kh + (j&7) + 16*(j>>3), n = m
#pragma unroll
    for (int j = 0; j < 16; ++j)
      B[j] = (_Float16)W2[(kh * 8 + (j & 7) + 16 * (j >> 3)) * 64 + t * 16 + m];

    v8f c = {};
    c = __builtin_amdgcn_wmma_f32_16x16x32_f16(false, A, false, B,
                                               (short)0, c, false, false);
    int   n    = t * 16 + m;
    float bias = b2[n];
    float wl   = Wl[n];
#pragma unroll
    for (int r = 0; r < 8; ++r)                  // c[r] -> (M = r+8*kh, N = n)
      p[r] += fmaxf(c[r] + bias, 0.0f) * wl;
  }

  // sum over the 16 lanes of each half (covers all 64 output columns per row)
#pragma unroll
  for (int off = 1; off < 16; off <<= 1)
#pragma unroll
    for (int r = 0; r < 8; ++r)
      p[r] += __shfl_xor(p[r], off, 16);

  if (m == 0) {
#pragma unroll
    for (int r = 0; r < 8; ++r) {
      int nd = wave * 16 + r + 8 * kh;
      if (nd < N) atomicAdd(&gsum[batch[nd]], p[r]);
    }
  }
}

__global__ void k_final(const float* __restrict__ gsum, const float* __restrict__ gcnt,
                        const float* __restrict__ bl, float* __restrict__ out, int G) {
  int g = blockIdx.x * blockDim.x + threadIdx.x;
  if (g < G) out[g] = gsum[g] / fmaxf(gcnt[g], 1.0f) + bl[0];
}

extern "C" void kernel_launch(void* const* d_in, const int* in_sizes, int n_in,
                              void* d_out, int out_size, void* d_ws, size_t ws_size,
                              hipStream_t stream) {
  const float* x     = (const float*)d_in[0];
  const int*   ei    = (const int*)d_in[1];
  const int*   batch = (const int*)d_in[2];
  const float* W1    = (const float*)d_in[3];
  const float* b1    = (const float*)d_in[4];
  const float* W2    = (const float*)d_in[5];
  const float* b2    = (const float*)d_in[6];
  const float* Wl    = (const float*)d_in[7];
  const float* bl    = (const float*)d_in[8];

  const int N = in_sizes[0];
  const int E = in_sizes[1] / 2;
  const int G = out_size;
  const int* src = ei;
  const int* dst = ei + E;

  float* ws     = (float*)d_ws;
  float* dinv   = ws;
  float* xs     = ws + N;                  // deg during phase 1
  float* asum   = ws + 2 * (size_t)N;
  float* gsum   = ws + 3 * (size_t)N;
  float* gcnt   = gsum + NGRAPH_MAX;
  float* h1s    = gcnt + NGRAPH_MAX;
  float* aggsum = h1s + 32 * (size_t)N;

  const int B = 256;
  k_init <<<(N + B - 1) / B, B, 0, stream>>>(xs, asum, gsum, gcnt, N);
  k_deg  <<<(E + B - 1) / B, B, 0, stream>>>(dst, xs, E);
  k_dinv <<<(N + B - 1) / B, B, 0, stream>>>(x, batch, xs, dinv, gcnt, N);
  k_edge1<<<(E + B - 1) / B, B, 0, stream>>>(src, dst, xs, asum, E);
  k_node1<<<((size_t)N * 32 + B - 1) / B, B, 0, stream>>>(asum, xs, dinv, W1, b1,
                                                          h1s, aggsum, N);
  int e32 = E * 32;
  k_edge2<<<(e32 + B - 1) / B, B, 0, stream>>>(src, dst, h1s, aggsum, e32);

  int ntiles  = (N + 15) / 16;
  int threads = ntiles * 32;
  k_gemm_pool<<<(threads + B - 1) / B, B, 0, stream>>>(aggsum, dinv, W2, b2, Wl,
                                                       batch, gsum, N);
  k_final<<<(G + B - 1) / B, B, 0, stream>>>(gsum, gcnt, bl, (float*)d_out, G);
}
